// DinoV2BackboneWrapper_37417755083337
// MI455X (gfx1250) — matrix operand order repaired and checked by reading, not compiled, not run on hardware
//
#include <hip/hip_runtime.h>
#include <math.h>

#define LNUM 12
#define CDIM 768
#define HNUM 12
#define DHD  64
#define SNUM 2
#define NTOK 577
#define MTOT 1154
#define MPAD 1168   /* 73*16 */
#define TILM 73
#define TILM4 19    /* ceil(73/4) macro M tiles (64 rows each) */
#define C4   3072
#define VW   1184   /* vT row length (2*592), 16-aligned per sequence */
#define VSEQ 592

typedef float  v8f   __attribute__((ext_vector_type(8)));
typedef __bf16 v16bf __attribute__((ext_vector_type(16)));
typedef __bf16 bf8   __attribute__((ext_vector_type(8)));

__device__ __forceinline__ v16bf cat8(bf8 lo, bf8 hi){
  return __builtin_shufflevector(lo, hi, 0,1,2,3,4,5,6,7,8,9,10,11,12,13,14,15);
}
__device__ __forceinline__ v8f wmma_bf16(v16bf a, v16bf b, v8f c){
  return __builtin_amdgcn_wmma_f32_16x16x32_bf16(false, a, false, b, (short)0, c, false, false);
}
__device__ __forceinline__ float wred_sum32(float v){
  #pragma unroll
  for (int o = 16; o >= 1; o >>= 1) v += __shfl_xor(v, o, 32);
  return v;
}
__device__ __forceinline__ float wred_sum16(float v){   // half-wave (lanes 0-15 / 16-31 separate)
  #pragma unroll
  for (int o = 1; o < 16; o <<= 1) v += __shfl_xor(v, o, 32);
  return v;
}
__device__ __forceinline__ float wred_max16(float v){
  #pragma unroll
  for (int o = 1; o < 16; o <<= 1) v = fmaxf(v, __shfl_xor(v, o, 32));
  return v;
}

// ---------------- fp32 -> bf16 conversion ----------------
__global__ __launch_bounds__(256) void f2bf_kernel(const float* __restrict__ s,
                                                   __bf16* __restrict__ d, int n){
  int t = blockIdx.x * 256 + threadIdx.x;
  if (t < n) d[t] = (__bf16)s[t];
}

// ---------------- camera token injection (layer 0) ----------------
__global__ __launch_bounds__(256) void cam_inject_kernel(float* __restrict__ x,
                                                         const float* __restrict__ cam){
  int t = blockIdx.x * 256 + threadIdx.x;
  if (t < SNUM * CDIM) {
    int s = t / CDIM, c = t % CDIM;
    x[(size_t)(s * NTOK) * CDIM + c] = cam[t];
  }
}

// ---------------- zero padded q/k rows (once per launch) ----------------
__global__ __launch_bounds__(256) void zeropad_kernel(__bf16* __restrict__ qb,
                                                      __bf16* __restrict__ kb){
  int t = blockIdx.x * 256 + threadIdx.x;
  if (t < (MPAD - MTOT) * CDIM) {
    qb[(size_t)MTOT * CDIM + t] = (__bf16)0.0f;
    kb[(size_t)MTOT * CDIM + t] = (__bf16)0.0f;
  }
}

// ---------------- LayerNorm -> bf16 (one wave per token) ----------------
__global__ __launch_bounds__(256) void ln_kernel(const float* __restrict__ x,
                                                 const float* __restrict__ w,
                                                 const float* __restrict__ b,
                                                 __bf16* __restrict__ h, int M){
  int wid = (blockIdx.x * blockDim.x + threadIdx.x) >> 5;
  if (wid >= M) return;
  int lane = threadIdx.x & 31;
  const float* xr = x + (size_t)wid * CDIM;
  float v[24]; float sm = 0.f, ssq = 0.f;
  #pragma unroll
  for (int i = 0; i < 24; ++i) { float t = xr[lane + i * 32]; v[i] = t; sm += t; ssq += t * t; }
  sm = wred_sum32(sm); ssq = wred_sum32(ssq);
  float mean = sm * (1.0f / CDIM);
  float var  = ssq * (1.0f / CDIM) - mean * mean;
  float inv  = rsqrtf(var + 1e-6f);
  #pragma unroll
  for (int i = 0; i < 24; ++i) {
    int c = lane + i * 32;
    h[(size_t)wid * CDIM + c] = (__bf16)((v[i] - mean) * inv * w[c] + b[c]);
  }
}

// ---------------- WMMA GEMM, 64x32 macro-tile per wave (4M x 2N register blocking) ----
// out[m][n] = sum_k A[m][k]*W[n][k]
// MODE 0: outF = acc+bias (qkv)    MODE 1: X += gamma*(acc+bias) (proj/fc2)
// MODE 2: outB = bf16(gelu(acc+bias)) (fc1)
template<int MODE>
__global__ __launch_bounds__(256) void gemm_tile(const __bf16* __restrict__ A,
                                                 const __bf16* __restrict__ W,
                                                 const float* __restrict__ bias,
                                                 const float* __restrict__ gamma,
                                                 float* __restrict__ Xr,
                                                 float* __restrict__ outF,
                                                 __bf16* __restrict__ outB,
                                                 int M, int Nn, int K, int tilesN2){
  int wid = (blockIdx.x * blockDim.x + threadIdx.x) >> 5;
  if (wid >= TILM4 * tilesN2) return;
  int tm = (wid / tilesN2) * 64, tn = (wid % tilesN2) * 32;
  int lane = threadIdx.x & 31, hi = lane >> 4, r = lane & 15;
  const __bf16* Ap[4];
  #pragma unroll
  for (int ms = 0; ms < 4; ++ms) {
    int row = tm + ms * 16 + r; if (row > MPAD - 1) row = MPAD - 1;   // clamp: pad-safe loads
    Ap[ms] = A + (size_t)row * K;
  }
  const __bf16* Wp0 = W + (size_t)(tn + r) * K;
  const __bf16* Wp1 = W + (size_t)(tn + 16 + r) * K;
  v8f acc[8] = {};
  for (int k0 = 0; k0 < K; k0 += 32) {
    __builtin_prefetch(Wp0 + k0 + 128, 0, 0);   // global_prefetch next weight lines
    __builtin_prefetch(Wp1 + k0 + 128, 0, 0);
    v16bf af[4];
    #pragma unroll
    for (int ms = 0; ms < 4; ++ms)
      af[ms] = cat8(*(const bf8*)(Ap[ms] + k0 + hi * 16),
                    *(const bf8*)(Ap[ms] + k0 + hi * 16 + 8));
    v16bf b0 = cat8(*(const bf8*)(Wp0 + k0 + hi * 16),
                    *(const bf8*)(Wp0 + k0 + hi * 16 + 8));
    v16bf b1 = cat8(*(const bf8*)(Wp1 + k0 + hi * 16),
                    *(const bf8*)(Wp1 + k0 + hi * 16 + 8));
    #pragma unroll
    for (int ms = 0; ms < 4; ++ms) {
      acc[ms * 2 + 0] = wmma_bf16(af[ms], b0, acc[ms * 2 + 0]);
      acc[ms * 2 + 1] = wmma_bf16(af[ms], b1, acc[ms * 2 + 1]);
    }
  }
  #pragma unroll
  for (int ms = 0; ms < 4; ++ms) {
    #pragma unroll
    for (int ns = 0; ns < 2; ++ns) {
      int col = tn + ns * 16 + r;
      #pragma unroll
      for (int j = 0; j < 8; ++j) {
        int row = tm + ms * 16 + j + hi * 8;
        if (row < M) {
          float t = acc[ms * 2 + ns][j] + bias[col];
          if (MODE == 0) {
            outF[(size_t)row * Nn + col] = t;
          } else if (MODE == 1) {
            Xr[(size_t)row * CDIM + col] += gamma[col] * t;
          } else {
            float g = 0.5f * t * (1.0f + erff(t * 0.70710678118654752f));
            outB[(size_t)row * Nn + col] = (__bf16)g;
          }
        }
      }
    }
  }
}

// ---------------- qkv post: RMS norm + RoPE, write bf16 q/k and transposed v ----------------
__global__ __launch_bounds__(256) void qkvpost_kernel(const float* __restrict__ qkv,
                                                      const float* __restrict__ qn,
                                                      const float* __restrict__ kn,
                                                      const float* __restrict__ cosT,
                                                      const float* __restrict__ sinT,
                                                      int isLocal,
                                                      __bf16* __restrict__ qb,
                                                      __bf16* __restrict__ kb,
                                                      __bf16* __restrict__ vT, int M){
  int wid = (blockIdx.x * blockDim.x + threadIdx.x) >> 5;
  if (wid >= M * HNUM) return;
  int m = wid / HNUM, h = wid % HNUM;
  int lane = threadIdx.x & 31;
  const float* base = qkv + (size_t)m * (3 * CDIM) + h * DHD;
  int d0 = lane * 2, d1 = d0 + 1;
  float q0 = base[d0],            q1 = base[d1];
  float k0 = base[CDIM + d0],     k1 = base[CDIM + d1];
  float v0 = base[2 * CDIM + d0], v1 = base[2 * CDIM + d1];
  float sq = wred_sum32(q0 * q0 + q1 * q1);
  float sk = wred_sum32(k0 * k0 + k1 * k1);
  float qi = rsqrtf(sq * (1.0f / DHD) + 1e-6f);
  float ki = rsqrtf(sk * (1.0f / DHD) + 1e-6f);
  q0 *= qi * qn[d0]; q1 *= qi * qn[d1];
  k0 *= ki * kn[d0]; k1 *= ki * kn[d1];
  // RoPE: out[d] = t[d]*cos[d] + s*t[d^16]*sin[d], s = (d&16)? +1 : -1 ; partner lane = lane^8
  float qp0 = __shfl_xor(q0, 8, 32), qp1 = __shfl_xor(q1, 8, 32);
  float kp0 = __shfl_xor(k0, 8, 32), kp1 = __shfl_xor(k1, 8, 32);
  float sg = (d0 & 16) ? 1.0f : -1.0f;
  int pos = isLocal ? (m % NTOK) : m;
  const float* cr = cosT + (size_t)pos * DHD;
  const float* sr = sinT + (size_t)pos * DHD;
  float qr0 = q0 * cr[d0] + sg * qp0 * sr[d0];
  float qr1 = q1 * cr[d1] + sg * qp1 * sr[d1];
  float kr0 = k0 * cr[d0] + sg * kp0 * sr[d0];
  float kr1 = k1 * cr[d1] + sg * kp1 * sr[d1];
  size_t qoff = (size_t)m * CDIM + h * DHD;
  qb[qoff + d0] = (__bf16)qr0; qb[qoff + d1] = (__bf16)qr1;
  kb[qoff + d0] = (__bf16)kr0; kb[qoff + d1] = (__bf16)kr1;
  int vidx = isLocal ? ((m / NTOK) * VSEQ + (m % NTOK)) : m;
  vT[(size_t)(h * DHD + d0) * VW + vidx] = (__bf16)v0;
  vT[(size_t)(h * DHD + d1) * VW + vidx] = (__bf16)v1;
}

// ---------------- fused flash attention: one wave per (seq, head, 16-query tile) ----------------
__global__ __launch_bounds__(256) void attn_kernel(const __bf16* __restrict__ qb,
                                                   const __bf16* __restrict__ kb,
                                                   const __bf16* __restrict__ vT,
                                                   const float* __restrict__ kv, int kvStride,
                                                   __bf16* __restrict__ ob,
                                                   int nseq, int SL, int qtiles){
  __shared__ __align__(16) __bf16 lds[8 * 512];
  int wid = (blockIdx.x * blockDim.x + threadIdx.x) >> 5;
  if (wid >= nseq * HNUM * qtiles) return;
  int seq = wid / (HNUM * qtiles);
  int rem = wid % (HNUM * qtiles);
  int h = rem / qtiles, qt = rem % qtiles;
  int lane = threadIdx.x & 31, hi = lane >> 4, r = lane & 15;
  int m0 = seq * SL;
  int vbase = seq * VSEQ;
  int qrow = m0 + qt * 16 + r; if (qrow > MTOT - 1) qrow = MTOT - 1;
  const __bf16* Qp = qb + (size_t)qrow * CDIM + h * DHD;
  __bf16* ldsw = lds + ((threadIdx.x >> 5) * 512);
  float mrow[8], lrow[8];
  v8f O0 = {}, O1 = {}, O2 = {}, O3 = {};
  #pragma unroll
  for (int j = 0; j < 8; ++j) { mrow[j] = -3.0e38f; lrow[j] = 0.f; }
  int nkt = (SL + 31) >> 5;
  for (int kt = 0; kt < nkt; ++kt) {
    int kq0 = kt * 32 + r, kq1 = kq0 + 16;
    int krow0 = m0 + kq0; if (krow0 > MTOT - 1) krow0 = MTOT - 1;
    int krow1 = m0 + kq1; if (krow1 > MTOT - 1) krow1 = MTOT - 1;
    const __bf16* K0p = kb + (size_t)krow0 * CDIM + h * DHD;
    const __bf16* K1p = kb + (size_t)krow1 * CDIM + h * DHD;
    v8f S0 = {}, S1 = {};
    #pragma unroll
    for (int dd0 = 0; dd0 < DHD; dd0 += 32) {
      v16bf af = cat8(*(const bf8*)(Qp + dd0 + hi * 16),
                      *(const bf8*)(Qp + dd0 + hi * 16 + 8));
      v16bf b0 = cat8(*(const bf8*)(K0p + dd0 + hi * 16),
                      *(const bf8*)(K0p + dd0 + hi * 16 + 8));
      v16bf b1 = cat8(*(const bf8*)(K1p + dd0 + hi * 16),
                      *(const bf8*)(K1p + dd0 + hi * 16 + 8));
      S0 = wmma_bf16(af, b0, S0);
      S1 = wmma_bf16(af, b1, S1);
    }
    float bias0 = (kq0 < SL) ? (1.0f - kv[seq * kvStride + kq0]) * -10000.0f : -1.0e30f;
    float bias1 = (kq1 < SL) ? (1.0f - kv[seq * kvStride + kq1]) * -10000.0f : -1.0e30f;
    float p0[8], p1[8];
    #pragma unroll
    for (int j = 0; j < 8; ++j) {
      float s0v = S0[j] * 0.125f + bias0;
      float s1v = S1[j] * 0.125f + bias1;
      float mx = wred_max16(fmaxf(s0v, s1v));
      float nm = fmaxf(mrow[j], mx);
      float e0 = expf(s0v - nm), e1 = expf(s1v - nm);
      float sum = wred_sum16(e0 + e1);
      float scale = expf(mrow[j] - nm);
      lrow[j] = lrow[j] * scale + sum;
      mrow[j] = nm;
      O0[j] *= scale; O1[j] *= scale; O2[j] *= scale; O3[j] *= scale;
      p0[j] = e0; p1[j] = e1;
    }
    // stage P (16x32) in LDS: C-layout -> A-layout redistribution
    #pragma unroll
    for (int j = 0; j < 8; ++j) {
      int rw = (j + hi * 8) * 32;
      ldsw[rw + r]      = (__bf16)p0[j];
      ldsw[rw + 16 + r] = (__bf16)p1[j];
    }
    asm volatile("s_wait_dscnt 0" ::: "memory");
    v16bf pf = cat8(*(const bf8*)(ldsw + r * 32 + hi * 16),
                    *(const bf8*)(ldsw + r * 32 + hi * 16 + 8));
    int kbase = vbase + kt * 32 + hi * 16; if (kbase > VW - 16) kbase = VW - 16;
    #pragma unroll
    for (int c = 0; c < 4; ++c) {
      int ddv = c * 16 + r;
      const __bf16* vp = vT + (size_t)(h * DHD + ddv) * VW + kbase;
      v16bf bf_ = cat8(*(const bf8*)vp, *(const bf8*)(vp + 8));
      if (c == 0) O0 = wmma_bf16(pf, bf_, O0);
      else if (c == 1) O1 = wmma_bf16(pf, bf_, O1);
      else if (c == 2) O2 = wmma_bf16(pf, bf_, O2);
      else O3 = wmma_bf16(pf, bf_, O3);
    }
  }
  #pragma unroll
  for (int j = 0; j < 8; ++j) {
    int qrel = qt * 16 + j + hi * 8;
    if (qrel < SL) {
      size_t rowoff = (size_t)(m0 + qrel) * CDIM + h * DHD;
      float inv = 1.0f / lrow[j];
      ob[rowoff + 0 + r]  = (__bf16)(O0[j] * inv);
      ob[rowoff + 16 + r] = (__bf16)(O1[j] * inv);
      ob[rowoff + 32 + r] = (__bf16)(O2[j] * inv);
      ob[rowoff + 48 + r] = (__bf16)(O3[j] * inv);
    }
  }
}

// ---------------- tap-out: concat(local_x, LN(x)) dropping token 0; optional cam out ----------------
__global__ __launch_bounds__(256) void out_kernel(const float* __restrict__ x,
                                                  const float* __restrict__ lx,
                                                  const float* __restrict__ fw,
                                                  const float* __restrict__ fb,
                                                  float* __restrict__ dst,
                                                  float* __restrict__ cam, int M){
  int wid = (blockIdx.x * blockDim.x + threadIdx.x) >> 5;
  if (wid >= M) return;
  int lane = threadIdx.x & 31;
  int s = wid / NTOK, n = wid % NTOK;
  const float* xr = x + (size_t)wid * CDIM;
  const float* lr = lx + (size_t)wid * CDIM;
  float v[24]; float sm = 0.f, ssq = 0.f;
  #pragma unroll
  for (int i = 0; i < 24; ++i) { float t = xr[lane + i * 32]; v[i] = t; sm += t; ssq += t * t; }
  sm = wred_sum32(sm); ssq = wred_sum32(ssq);
  float mean = sm * (1.0f / CDIM);
  float var  = ssq * (1.0f / CDIM) - mean * mean;
  float inv  = rsqrtf(var + 1e-6f);
  if (n > 0) {
    float* d = dst + ((size_t)(s * (NTOK - 1) + n - 1)) * (2 * CDIM);
    #pragma unroll
    for (int i = 0; i < 24; ++i) {
      int c = lane + i * 32;
      d[c]        = lr[c];
      d[CDIM + c] = (v[i] - mean) * inv * fw[c] + fb[c];
    }
  } else if (cam) {
    #pragma unroll
    for (int i = 0; i < 24; ++i) {
      int c = lane + i * 32;
      cam[s * 2 * CDIM + c]        = lr[c];
      cam[s * 2 * CDIM + CDIM + c] = xr[c];   // raw x, no final LN for cam token
    }
  }
}

// ======================= host orchestration =======================
extern "C" void kernel_launch(void* const* d_in, const int* in_sizes, int n_in,
                              void* d_out, int out_size, void* d_ws, size_t ws_size,
                              hipStream_t stream) {
  (void)in_sizes; (void)n_in; (void)out_size; (void)ws_size;
  const float* in_x   = (const float*)d_in[0];
  const float* cosL   = (const float*)d_in[1];
  const float* sinL   = (const float*)d_in[2];
  const float* cosG   = (const float*)d_in[3];
  const float* sinG   = (const float*)d_in[4];
  const float* kvL    = (const float*)d_in[5];
  const float* kvG    = (const float*)d_in[6];
  const float* camTok = (const float*)d_in[7];
  const float* qkv_w  = (const float*)d_in[8];
  const float* qkv_b  = (const float*)d_in[9];
  const float* qn_w   = (const float*)d_in[10];
  const float* kn_w   = (const float*)d_in[11];
  const float* proj_w = (const float*)d_in[12];
  const float* proj_b = (const float*)d_in[13];
  const float* g1     = (const float*)d_in[14];
  const float* g2     = (const float*)d_in[15];
  const float* n1w    = (const float*)d_in[16];
  const float* n1b    = (const float*)d_in[17];
  const float* n2w    = (const float*)d_in[18];
  const float* n2b    = (const float*)d_in[19];
  const float* fc1w   = (const float*)d_in[20];
  const float* fc1b   = (const float*)d_in[21];
  const float* fc2w   = (const float*)d_in[22];
  const float* fc2b   = (const float*)d_in[23];
  const float* fnw    = (const float*)d_in[24];
  const float* fnb    = (const float*)d_in[25];
  float* out = (float*)d_out;

  char* base = (char*)d_ws;
  size_t off = 0;
  auto take = [&](size_t bytes) -> char* {
    char* p = base + off;
    off += (bytes + 255) & ~(size_t)255;
    return p;
  };
  __bf16* Wqkv = (__bf16*)take((size_t)LNUM * 3 * CDIM * CDIM * 2);
  __bf16* Wprj = (__bf16*)take((size_t)LNUM * CDIM * CDIM * 2);
  __bf16* Wf1  = (__bf16*)take((size_t)LNUM * C4 * CDIM * 2);
  __bf16* Wf2  = (__bf16*)take((size_t)LNUM * CDIM * C4 * 2);
  float*  X    = (float*) take((size_t)MPAD * CDIM * 4);
  float*  LX   = (float*) take((size_t)MPAD * CDIM * 4);
  __bf16* Hb   = (__bf16*)take((size_t)MPAD * CDIM * 2);
  float*  QKV  = (float*) take((size_t)MPAD * 3 * CDIM * 4);
  __bf16* Qb   = (__bf16*)take((size_t)MPAD * CDIM * 2);
  __bf16* Kb   = (__bf16*)take((size_t)MPAD * CDIM * 2);
  __bf16* Vt   = (__bf16*)take((size_t)CDIM * VW * 2);
  __bf16* Ob   = (__bf16*)take((size_t)MPAD * CDIM * 2);
  __bf16* Mb   = (__bf16*)take((size_t)MPAD * C4 * 2);

  auto conv = [&](const float* s, __bf16* d, size_t n) {
    f2bf_kernel<<<(int)((n + 255) / 256), 256, 0, stream>>>(s, d, (int)n);
  };
  conv(qkv_w, Wqkv, (size_t)LNUM * 3 * CDIM * CDIM);
  conv(proj_w, Wprj, (size_t)LNUM * CDIM * CDIM);
  conv(fc1w,  Wf1,  (size_t)LNUM * C4 * CDIM);
  conv(fc2w,  Wf2,  (size_t)LNUM * CDIM * C4);

  hipMemcpyAsync(X, in_x, (size_t)MTOT * CDIM * 4, hipMemcpyDeviceToDevice, stream);
  zeropad_kernel<<<((MPAD - MTOT) * CDIM + 255) / 256, 256, 0, stream>>>(Qb, Kb);

  const int OUTL[4] = {2, 5, 8, 11};
  for (int i = 0; i < LNUM; ++i) {
    bool isGlobal = (i & 1);
    if (i == 0)
      cam_inject_kernel<<<(SNUM * CDIM + 255) / 256, 256, 0, stream>>>(X, camTok);

    int lnBlocks = (MTOT * 32 + 255) / 256;
    ln_kernel<<<lnBlocks, 256, 0, stream>>>(X, n1w + i * CDIM, n1b + i * CDIM, Hb, MTOT);

    { // qkv GEMM: M x 2304, K=768
      int tilesN2 = (3 * CDIM) / 32, waves = TILM4 * tilesN2;
      gemm_tile<0><<<(waves * 32 + 255) / 256, 256, 0, stream>>>(
          Hb, Wqkv + (size_t)i * 3 * CDIM * CDIM, qkv_b + i * 3 * CDIM,
          nullptr, nullptr, QKV, nullptr, MTOT, 3 * CDIM, CDIM, tilesN2);
    }

    hipMemsetAsync(Vt, 0, (size_t)CDIM * VW * 2, stream);
    {
      int waves = MTOT * HNUM;
      qkvpost_kernel<<<(waves * 32 + 255) / 256, 256, 0, stream>>>(
          QKV, qn_w + i * DHD, kn_w + i * DHD,
          isGlobal ? cosG : cosL, isGlobal ? sinG : sinL, isGlobal ? 0 : 1,
          Qb, Kb, Vt, MTOT);
    }

    {
      int nseq = isGlobal ? 1 : SNUM;
      int SL   = isGlobal ? MTOT : NTOK;
      int qtiles = (SL + 15) / 16;
      int waves = nseq * HNUM * qtiles;
      attn_kernel<<<(waves * 32 + 255) / 256, 256, 0, stream>>>(
          Qb, Kb, Vt, isGlobal ? kvG : kvL, SL, Ob, nseq, SL, qtiles);
    }

    { // proj GEMM + residual: x += g1*(o@W + b)
      int tilesN2 = CDIM / 32, waves = TILM4 * tilesN2;
      gemm_tile<1><<<(waves * 32 + 255) / 256, 256, 0, stream>>>(
          Ob, Wprj + (size_t)i * CDIM * CDIM, proj_b + i * CDIM,
          g1 + i * CDIM, X, nullptr, nullptr, MTOT, CDIM, CDIM, tilesN2);
    }

    ln_kernel<<<lnBlocks, 256, 0, stream>>>(X, n2w + i * CDIM, n2b + i * CDIM, Hb, MTOT);

    { // fc1 GEMM + GELU -> bf16
      int tilesN2 = C4 / 32, waves = TILM4 * tilesN2;
      gemm_tile<2><<<(waves * 32 + 255) / 256, 256, 0, stream>>>(
          Hb, Wf1 + (size_t)i * C4 * CDIM, fc1b + i * C4,
          nullptr, nullptr, nullptr, Mb, MTOT, C4, CDIM, tilesN2);
    }
    { // fc2 GEMM + residual: x += g2*(m@W + b)
      int tilesN2 = CDIM / 32, waves = TILM4 * tilesN2;
      gemm_tile<1><<<(waves * 32 + 255) / 256, 256, 0, stream>>>(
          Mb, Wf2 + (size_t)i * CDIM * C4, fc2b + i * CDIM,
          g2 + i * CDIM, X, nullptr, nullptr, MTOT, CDIM, C4, tilesN2);
    }

    if (!isGlobal)
      hipMemcpyAsync(LX, X, (size_t)MTOT * CDIM * 4, hipMemcpyDeviceToDevice, stream);

    for (int p = 0; p < 4; ++p) {
      if (OUTL[p] == i) {
        size_t slotSz = (size_t)SNUM * (NTOK - 1) * 2 * CDIM;
        float* dst = out + (size_t)p * slotSz;
        float* cam = (p == 3) ? (out + 4 * slotSz) : nullptr;
        out_kernel<<<lnBlocks, 256, 0, stream>>>(X, LX, fnw, fnb, dst, cam, MTOT);
      }
    }
  }
}
